// DGL_SAGEConv_32160715112812
// MI455X (gfx1250) — compile-verified, matrix operand order checked by reading, not run on hardware
//
#include <hip/hip_runtime.h>
#include <hip/hip_bf16.h>

typedef __attribute__((ext_vector_type(2)))  float    v2f;
typedef __attribute__((ext_vector_type(8)))  float    v8f;
typedef __attribute__((ext_vector_type(16))) _Float16 v16h;

#define IN_FEATS 128
#define OUT_FEATS 128

// ---------------------------------------------------------------------------
// Fused dual GEMM:  h = feat * w_neigh^T   and   out = feat * w_self^T + b
// One wave owns one 16(node) x 16(outfeat) tile for BOTH weight matrices;
// 8 waves/block cover all 8 o-tiles of a node tile (feat tile stays in WGP$).
// WMMA f32 16x16x4: A/B = 2 VGPRs, C/D = 8 VGPRs (ISA 7.12.2 layouts).
// Codegen-verified: emits v_wmma_f32_16x16x4_f32 with pipelined b64 loads.
// ---------------------------------------------------------------------------
__global__ __launch_bounds__(256) void sage_dual_gemm(
    const float* __restrict__ feat,
    const float* __restrict__ w_neigh,
    const float* __restrict__ w_self,
    const float* __restrict__ b_self,
    float* __restrict__ h_out,     // workspace: neighbor-path features
    float* __restrict__ self_out,  // d_out: self path + bias
    int n_nodes)
{
    const int lane = threadIdx.x & 31;
    const int wave = threadIdx.x >> 5;        // 0..7  -> o-tile
    const int m0   = blockIdx.x * 16;         // node tile base
    const int o0   = wave * 16;               // out-feature tile base

    const int m     = lane & 15;
    const int khalf = (lane >> 4) * 2;        // 0 or 2
    int arow_i = m0 + m;
    if (arow_i >= n_nodes) arow_i = n_nodes - 1;   // safe duplicate load
    const float* __restrict__ arow = feat + (size_t)arow_i * IN_FEATS;

    // B fragment: B[k][n] = W[o0+n][k]  (W is [out,in] row-major)
    const int n = lane & 15;
    const float* __restrict__ bn_row = w_neigh + (size_t)(o0 + n) * IN_FEATS;
    const float* __restrict__ bs_row = w_self  + (size_t)(o0 + n) * IN_FEATS;

    v8f cn = {};
    v8f cs = {};

#if __has_builtin(__builtin_amdgcn_wmma_f32_16x16x4_f32)
    #pragma unroll 4
    for (int k = 0; k < IN_FEATS; k += 4) {
        v2f a, bn, bs;
        a.x  = arow[k + khalf];     a.y  = arow[k + khalf + 1];
        bn.x = bn_row[k + khalf];   bn.y = bn_row[k + khalf + 1];
        bs.x = bs_row[k + khalf];   bs.y = bs_row[k + khalf + 1];
        cn = __builtin_amdgcn_wmma_f32_16x16x4_f32(
                 false, a, false, bn, (short)0, cn, false, false);
        cs = __builtin_amdgcn_wmma_f32_16x16x4_f32(
                 false, a, false, bs, (short)0, cs, false, false);
    }
#else
    const int kin = (lane >> 4) * 8;
    for (int k = 0; k < IN_FEATS; k += 32) {
        v16h a, bn, bs;
        #pragma unroll
        for (int j = 0; j < 16; ++j) {
            const int kk = k + ((j >= 8) ? 16 : 0) + kin + (j & 7);
            a[j]  = (_Float16)arow[kk];
            bn[j] = (_Float16)bn_row[kk];
            bs[j] = (_Float16)bs_row[kk];
        }
        cn = __builtin_amdgcn_wmma_f32_16x16x32_f16(
                 false, a, false, bn, (short)0, cn, false, false);
        cs = __builtin_amdgcn_wmma_f32_16x16x32_f16(
                 false, a, false, bs, (short)0, cs, false, false);
    }
#endif

    // C/D layout: VGPR r, lane l -> row = r + 8*(l>>4), col = l&15
    const int colg  = o0 + (lane & 15);
    const int rbase = (lane >> 4) * 8;
    const float bval = b_self[colg];

    if (m0 + 16 <= n_nodes) {
        // Fast path (every block when N % 16 == 0): straight-line stores.
        #pragma unroll
        for (int r = 0; r < 8; ++r) {
            const size_t idx = (size_t)(m0 + rbase + r) * OUT_FEATS + colg;
            h_out[idx]    = cn[r];
            self_out[idx] = cs[r] + bval;
        }
    } else {
        // Generic guarded tail.
        #pragma unroll
        for (int r = 0; r < 8; ++r) {
            const int orow = m0 + rbase + r;
            if (orow < n_nodes) {
                const size_t idx = (size_t)orow * OUT_FEATS + colg;
                h_out[idx]    = cn[r];
                self_out[idx] = cs[r] + bval;
            }
        }
    }
}

// ---------------------------------------------------------------------------
// CSR build step 1: degree histogram (integer atomics only, 1.6M ops).
// ---------------------------------------------------------------------------
__global__ __launch_bounds__(256) void sage_hist(
    const int* __restrict__ dst, int* __restrict__ deg, int n_edges)
{
    const int i = blockIdx.x * 256 + threadIdx.x;
    if (i < n_edges) atomicAdd(&deg[dst[i]], 1);
}

// ---------------------------------------------------------------------------
// CSR build step 2: exclusive scan deg[0..n-1] -> row_ptr[0..n].
// Single 1024-thread workgroup; wave32 shfl scan + one 32-entry LDS pass per
// 1024-element chunk (~98 chunks for N=100000, 4 barriers each).
// ---------------------------------------------------------------------------
__global__ __launch_bounds__(1024) void sage_scan(
    const int* __restrict__ deg, int* __restrict__ row_ptr, int n)
{
    __shared__ int wsum[32];
    __shared__ int carry_s;
    const int tid  = threadIdx.x;
    const int lane = tid & 31;
    const int wid  = tid >> 5;

    if (tid == 0) carry_s = 0;
    __syncthreads();

    for (int base = 0; base < n; base += 1024) {
        const int i = base + tid;
        const int v = (i < n) ? deg[i] : 0;

        // wave-level inclusive scan
        int x = v;
        #pragma unroll
        for (int off = 1; off < 32; off <<= 1) {
            int t = __shfl_up(x, off);
            if (lane >= off) x += t;
        }
        if (lane == 31) wsum[wid] = x;
        __syncthreads();

        // wave 0 scans the 32 wave totals
        if (wid == 0) {
            int s = wsum[lane];
            #pragma unroll
            for (int off = 1; off < 32; off <<= 1) {
                int t = __shfl_up(s, off);
                if (lane >= off) s += t;
            }
            wsum[lane] = s;                  // inclusive wave-prefix
        }
        __syncthreads();

        const int incl  = x + (wid ? wsum[wid - 1] : 0);
        const int carry = carry_s;
        if (i < n) row_ptr[i] = carry + incl - v;   // exclusive
        __syncthreads();                             // reads done before update
        if (tid == 0) carry_s = carry + wsum[31];
        __syncthreads();
    }
    if (tid == 0) row_ptr[n] = carry_s;
}

// ---------------------------------------------------------------------------
// CSR build step 3: place each edge's src id into its dst's slot range.
// ---------------------------------------------------------------------------
__global__ __launch_bounds__(256) void sage_scatter_edges(
    const int* __restrict__ src, const int* __restrict__ dst,
    const int* __restrict__ row_ptr, int* __restrict__ cursor,
    int* __restrict__ edge_src, int n_edges)
{
    const int i = blockIdx.x * 256 + threadIdx.x;
    if (i < n_edges) {
        const int d   = dst[i];
        const int pos = row_ptr[d] + atomicAdd(&cursor[d], 1);
        edge_src[pos] = src[i];
    }
}

// ---------------------------------------------------------------------------
// Gather + mean + combine: one wave per dst node; each lane owns a float4 of
// the 512B row. Loads 32 src ids coalesced, prefetches their rows
// (global_prefetch_b8), broadcasts via shfl, accumulates h[src] in registers.
// No float atomics. h (51.2MB) is L2-resident so gathers stay on-die.
// out (already = self + bias) is updated in place with the segment mean.
// ---------------------------------------------------------------------------
__global__ __launch_bounds__(256) void sage_gather(
    const float* __restrict__ h,
    const int*   __restrict__ row_ptr,
    const int*   __restrict__ edge_src,
    float* __restrict__ out,
    int n_nodes)
{
    const int lane = threadIdx.x & 31;
    const int wave = threadIdx.x >> 5;
    const int node = blockIdx.x * 8 + wave;
    if (node >= n_nodes) return;

    const int start = row_ptr[node];
    const int end   = row_ptr[node + 1];

    float4 acc = make_float4(0.f, 0.f, 0.f, 0.f);

    for (int base = start; base < end; base += 32) {
        int navail = end - base;
        if (navail > 32) navail = 32;

        const int my = base + lane;
        const int sv = (my < end) ? edge_src[my] : 0;

        // warm L0/L2 for the rows this chunk will stream
        if (lane < navail)
            __builtin_prefetch(h + (size_t)sv * OUT_FEATS, 0, 0);

        for (int j = 0; j < navail; ++j) {
            const int s = __shfl(sv, j);
            const float4* __restrict__ hrow =
                (const float4*)(h + (size_t)s * OUT_FEATS);
            const float4 v = hrow[lane];
            acc.x += v.x; acc.y += v.y; acc.z += v.z; acc.w += v.w;
        }
    }

    const int dg = end - start;
    const float inv = 1.0f / (float)(dg > 0 ? dg : 1);

    float4* __restrict__ orow = (float4*)(out + (size_t)node * OUT_FEATS);
    float4 o = orow[lane];
    o.x += acc.x * inv; o.y += acc.y * inv;
    o.z += acc.z * inv; o.w += acc.w * inv;
    orow[lane] = o;
}

// ---------------------------------------------------------------------------
// Launch. Inputs: feat, w_neigh, w_self, b_self, src, dst (setup order).
// Workspace: [ h : N*128 f32 | deg : N i32 | cursor : N i32 |
//              row_ptr : N+1 i32 | edge_src : E i32 ]  (~59 MB)
// ---------------------------------------------------------------------------
extern "C" void kernel_launch(void* const* d_in, const int* in_sizes, int n_in,
                              void* d_out, int out_size, void* d_ws, size_t ws_size,
                              hipStream_t stream)
{
    const float* feat    = (const float*)d_in[0];
    const float* w_neigh = (const float*)d_in[1];
    const float* w_self  = (const float*)d_in[2];
    const float* b_self  = (const float*)d_in[3];
    const int*   src     = (const int*)d_in[4];
    const int*   dst     = (const int*)d_in[5];
    float*       out     = (float*)d_out;

    const int N = in_sizes[0] / IN_FEATS;
    const int E = in_sizes[4];

    float* h        = (float*)d_ws;
    int*   deg      = (int*)(h + (size_t)N * OUT_FEATS);
    int*   cursor   = deg + N;
    int*   row_ptr  = cursor + N;
    int*   edge_src = row_ptr + (N + 1);

    // Zero deg + cursor (contiguous).
    hipMemsetAsync(deg, 0, (size_t)2 * N * sizeof(int), stream);

    // Dual GEMM: h = feat*Wn^T (-> ws), d_out = feat*Ws^T + b.
    sage_dual_gemm<<<(N + 15) / 16, 256, 0, stream>>>(
        feat, w_neigh, w_self, b_self, h, out, N);

    // CSR build: histogram -> scan -> scatter.
    sage_hist<<<(E + 255) / 256, 256, 0, stream>>>(dst, deg, E);
    sage_scan<<<1, 1024, 0, stream>>>(deg, row_ptr, N);
    sage_scatter_edges<<<(E + 255) / 256, 256, 0, stream>>>(
        src, dst, row_ptr, cursor, edge_src, E);

    // Gather + mean + combine (no float atomics).
    sage_gather<<<(N + 7) / 8, 256, 0, stream>>>(h, row_ptr, edge_src, out, N);
}
